// KosmosTextAttention_88630945120584
// MI455X (gfx1250) — matrix operand order repaired and checked by reading, not compiled, not run on hardware
//
#include <hip/hip_runtime.h>
#include <hip/hip_bf16.h>

typedef __bf16 bf16_t;
typedef __attribute__((ext_vector_type(16))) __bf16 v16bf;
typedef __attribute__((ext_vector_type(8)))  __bf16 v8bf;
typedef __attribute__((ext_vector_type(8)))  float  v8f;
typedef unsigned int u32;
typedef u32 u32x4 __attribute__((ext_vector_type(4)));
typedef u32 u32x8 __attribute__((ext_vector_type(8)));

#define E_DIM 2048
#define S_DIM 2048
#define B_DIM 2
#define H_DIM 32
#define D_DIM 64
#define M_DIM (B_DIM * S_DIM)   /* 4096 token rows */
#define Q_SCALE 0.125f          /* D^-0.5 = 64^-0.5 */
#define NEG_MASK (-1e9f)

// ---------------- WMMA fragment helpers (CDNA5 16-bit layouts) ----------------
// A-matrix 16x32 (bf16): lane = {half,row}; per-lane 16 elems, K scattered:
//   K = (vg>>2)*16 + half*8 + (vg&3)*2 + (j&1)
static __device__ inline v16bf load_afrag(const bf16_t* __restrict__ tile, int ld, int lane) {
  const int row = lane & 15, half = lane >> 4;
  v16bf a;
#pragma unroll
  for (int j = 0; j < 16; ++j) {
    const int vg = j >> 1;
    const int k = ((vg >> 2) << 4) + (half << 3) + ((vg & 3) << 1) + (j & 1);
    a[j] = tile[row * ld + k];
  }
  return a;
}

// B-matrix 32x16 (bf16) from an [N][K] (n-major) tile: lane = {half,col};
//   element j -> (K = (j>>3)*16 + half*8 + (j&7), N = col)
static __device__ inline v16bf load_bfrag_nk(const bf16_t* __restrict__ tile, int ld, int lane) {
  const int col = lane & 15, half = lane >> 4;
  v16bf b;
#pragma unroll
  for (int j = 0; j < 16; ++j) b[j] = tile[col * ld + ((j >> 3) << 4) + (half << 3) + (j & 7)];
  return b;
}

static __device__ inline v8f wmma_bf16(v16bf a, v16bf b, v8f c) {
  return __builtin_amdgcn_wmma_f32_16x16x32_bf16(false, a, false, b, (short)0, c, false, false);
}

// ---------------- Tensor Data Mover: 2D tile -> LDS (CDNA5 ISA) ----
// Builds a user-mode D# (groups 0+1, 2D tensor, VADDR2/3 = NULL form) in SGPRs and
// issues tensor_load_to_lds. bf16 elements (data_size=1 -> 2 bytes). All operands
// must be wave-uniform (they are: block indices / loop counters / LDS offsets).
static __device__ inline void tdm_load_tile_2d(u32 lds_byte_addr,
                                               const bf16_t* gtile,
                                               u32 tensor_d0, u32 tensor_d1,
                                               u32 tile_d0, u32 tile_d1,
                                               u32 stride0_elems) {
  const unsigned long long ga = (unsigned long long)(size_t)gtile;
  u32x4 g0;
  g0[0] = 1u;                                                  // count=1, user desc
  g0[1] = lds_byte_addr;                                       // lds_addr
  g0[2] = (u32)ga;                                             // global_addr[31:0]
  g0[3] = ((u32)(ga >> 32) & 0x01FFFFFFu) | (2u << 30);        // ga[56:32] | type=2
  u32x8 g1;
  g1[0] = 1u << 16;                          // workgroup_mask=0, data_size=1 (2B)
  g1[1] = tensor_d0 << 16;                   // tensor_dim0[15:0] (barrier addr = 0)
  g1[2] = (tensor_d0 >> 16) | (tensor_d1 << 16);
  g1[3] = (tensor_d1 >> 16) | (tile_d0 << 16);
  g1[4] = tile_d1 & 0xFFFFu;                 // tile_dim1 (tile_dim2 = 0 -> 2D)
  g1[5] = stride0_elems;                     // tensor_dim0_stride[31:0]
  g1[6] = 0u;                                // stride0 hi, dim1_stride lo (unused, 2D)
  g1[7] = 0u;
  asm volatile("tensor_load_to_lds %0, %1" :: "s"(g0), "s"(g1) : "memory");
}

// ---------------- fp32 -> bf16 conversion ----------------
__global__ __launch_bounds__(256) void cvt_kernel(const float* __restrict__ in,
                                                  bf16_t* __restrict__ out, int n) {
  const int idx = (blockIdx.x * 256 + threadIdx.x) * 4;
  if (idx + 3 < n) {
    const float4 f = *(const float4*)&in[idx];
    out[idx + 0] = (bf16_t)f.x;
    out[idx + 1] = (bf16_t)f.y;
    out[idx + 2] = (bf16_t)f.z;
    out[idx + 3] = (bf16_t)f.w;
  }
}

// ---------------- Tiled bf16 WMMA GEMM, TDM-fed, double buffered ----------------
// out = (X @ W^T + bias) * scale.  128x128 macro-tile, BK=32, 8 waves.
// MODE 0/1: write bf16 to [B,H,S,D] (QKV); MODE 2: write fp32 to [B,S,E] (d_out)
template <int MODE>
__global__ __launch_bounds__(256) void gemm_kernel(const bf16_t* __restrict__ X,
                                                   const bf16_t* __restrict__ W,
                                                   const float* __restrict__ bias,
                                                   float scale, void* __restrict__ outv) {
  __shared__ __align__(16) bf16_t Atile[2][128 * 32];   // [m][k], double buffered
  __shared__ __align__(16) bf16_t Btile[2][128 * 32];   // [n][k], double buffered
  const int tid = threadIdx.x;
  const int lane = tid & 31, wave = tid >> 5;
  const int m0 = blockIdx.y * 128, n0 = blockIdx.x * 128;
  const int mw = wave & 3, nw = wave >> 2;   // wave -> rows mw*32..+31, cols nw*64..+63

  const u32 ldsA[2] = {(u32)(size_t)&Atile[0][0], (u32)(size_t)&Atile[1][0]};
  const u32 ldsB[2] = {(u32)(size_t)&Btile[0][0], (u32)(size_t)&Btile[1][0]};

  v8f acc[2][4];
  const v8f vzero = {};
#pragma unroll
  for (int s = 0; s < 2; ++s)
#pragma unroll
    for (int t = 0; t < 4; ++t) acc[s][t] = vzero;

  // preload K-step 0 into buffer 0 (wave 0 drives the TDM)
  if (wave == 0) {
    tdm_load_tile_2d(ldsA[0], X + (size_t)m0 * E_DIM, E_DIM, M_DIM, 32, 128, E_DIM);
    tdm_load_tile_2d(ldsB[0], W + (size_t)n0 * E_DIM, E_DIM, E_DIM, 32, 128, E_DIM);
  }

  const int nsteps = E_DIM / 32;   // 64
  for (int i = 0; i < nsteps; ++i) {
    const int buf = i & 1;
    if (wave == 0) {
      if (i + 1 < nsteps) {
        const int k1 = (i + 1) * 32;   // prefetch next K-step into the other buffer
        tdm_load_tile_2d(ldsA[buf ^ 1], X + (size_t)m0 * E_DIM + k1, E_DIM, M_DIM, 32, 128, E_DIM);
        tdm_load_tile_2d(ldsB[buf ^ 1], W + (size_t)n0 * E_DIM + k1, E_DIM, E_DIM, 32, 128, E_DIM);
        __builtin_amdgcn_s_wait_tensorcnt(2);  // TDM in-order per wave => step i landed
      } else {
        __builtin_amdgcn_s_wait_tensorcnt(0);
      }
    }
    __syncthreads();   // publish buffer `buf` to all waves

    const v16bf a0 = load_afrag(&Atile[buf][(mw * 32 + 0) * 32], 32, lane);
    const v16bf a1 = load_afrag(&Atile[buf][(mw * 32 + 16) * 32], 32, lane);
#pragma unroll
    for (int t = 0; t < 4; ++t) {
      const v16bf bfr = load_bfrag_nk(&Btile[buf][(nw * 64 + t * 16) * 32], 32, lane);
      acc[0][t] = wmma_bf16(a0, bfr, acc[0][t]);
      acc[1][t] = wmma_bf16(a1, bfr, acc[1][t]);
    }
    __syncthreads();   // everyone done reading before TDM overwrites this buffer
  }

  const int half = lane >> 4, col = lane & 15;
#pragma unroll
  for (int s = 0; s < 2; ++s) {
#pragma unroll
    for (int t = 0; t < 4; ++t) {
#pragma unroll
      for (int r = 0; r < 8; ++r) {
        const int m = m0 + mw * 32 + s * 16 + r + half * 8;  // C layout: M = vgpr + half*8
        const int n = n0 + nw * 64 + t * 16 + col;           //           N = lane&15
        const float val = (acc[s][t][r] + bias[n]) * scale;
        if (MODE == 2) {
          ((float*)outv)[(size_t)m * E_DIM + n] = val;
        } else {
          const int bb = m >> 11, ss = m & (S_DIM - 1);
          const int hh = n >> 6,  dd = n & 63;
          ((bf16_t*)outv)[((((size_t)bb * H_DIM + hh) * S_DIM + ss) << 6) + dd] = (bf16_t)val;
        }
      }
    }
  }
}

// ---------------- Flash attention (causal), bf16 WMMA, fp32 softmax ----------------
__global__ __launch_bounds__(256) void attn_kernel(const bf16_t* __restrict__ Q,
                                                   const bf16_t* __restrict__ K,
                                                   const bf16_t* __restrict__ V,
                                                   float* __restrict__ ctx) {
  __shared__ __align__(16) bf16_t Pbuf[8][16 * 32]; // per-wave P tile (row-major)
  __shared__ bf16_t VT[8][64 * 33];                 // per-wave V tile transposed [d][t]
  const int tid = threadIdx.x, lane = tid & 31, wave = tid >> 5;
  const int half = lane >> 4, col = lane & 15;
  const int hh = blockIdx.y, bb = blockIdx.z;
  const size_t base = (size_t)(bb * H_DIM + hh) * S_DIM * D_DIM;
  const int q0 = blockIdx.x * 128 + wave * 16;      // this wave's 16 query rows

  // Q A-fragments for K-steps 0/1 (held in registers for the whole kernel)
  const v16bf qa0 = load_afrag(&Q[base + (size_t)q0 * D_DIM + 0],  D_DIM, lane);
  const v16bf qa1 = load_afrag(&Q[base + (size_t)q0 * D_DIM + 32], D_DIM, lane);

  float mrun[8], lrun[8];
  v8f o[4];
  const v8f vzero = {};
#pragma unroll
  for (int r = 0; r < 8; ++r) { mrun[r] = -3.0e38f; lrun[r] = 0.f; }
#pragma unroll
  for (int dt = 0; dt < 4; ++dt) o[dt] = vzero;

  const int ntiles = (q0 + 16 + 31) / 32;           // causal key-tile bound
  for (int kt = 0; kt < ntiles; ++kt) {
    const int t0 = kt * 32;

    // stage V tile transposed into this wave's LDS region: VT[d][t_local]
#pragma unroll
    for (int c = 0; c < 8; ++c) {
      const v8bf vv = *(const v8bf*)&V[base + (size_t)(t0 + lane) * D_DIM + c * 8];
#pragma unroll
      for (int i = 0; i < 8; ++i) VT[wave][(c * 8 + i) * 33 + lane] = vv[i];
    }

    // S = Q K^T over 32 keys (2 N-tiles x 2 K-steps)
    v8f s0 = vzero, s1 = vzero;
    {
      v16bf bk;
      bk = load_bfrag_nk(&K[base + (size_t)(t0 + 0)  * D_DIM + 0],  D_DIM, lane); s0 = wmma_bf16(qa0, bk, s0);
      bk = load_bfrag_nk(&K[base + (size_t)(t0 + 0)  * D_DIM + 32], D_DIM, lane); s0 = wmma_bf16(qa1, bk, s0);
      bk = load_bfrag_nk(&K[base + (size_t)(t0 + 16) * D_DIM + 0],  D_DIM, lane); s1 = wmma_bf16(qa0, bk, s1);
      bk = load_bfrag_nk(&K[base + (size_t)(t0 + 16) * D_DIM + 32], D_DIM, lane); s1 = wmma_bf16(qa1, bk, s1);
    }

    // causal mask + online softmax (rows live across 16-lane groups)
#pragma unroll
    for (int r = 0; r < 8; ++r) {
      const int m = q0 + r + half * 8;
      if (t0 + col > m)      s0[r] += NEG_MASK;
      if (t0 + 16 + col > m) s1[r] += NEG_MASK;
      float mx = fmaxf(s0[r], s1[r]);
#pragma unroll
      for (int off = 1; off < 16; off <<= 1) mx = fmaxf(mx, __shfl_xor(mx, off, 32));
      const float mnew  = fmaxf(mrun[r], mx);
      const float alpha = __expf(mrun[r] - mnew);
      mrun[r] = mnew;
      const float p0 = __expf(s0[r] - mnew);
      const float p1 = __expf(s1[r] - mnew);
      float rs = p0 + p1;
#pragma unroll
      for (int off = 1; off < 16; off <<= 1) rs += __shfl_xor(rs, off, 32);
      lrun[r] = lrun[r] * alpha + rs;
#pragma unroll
      for (int dt = 0; dt < 4; ++dt) o[dt][r] *= alpha;
      Pbuf[wave][(r + half * 8) * 32 + col]      = (bf16_t)p0;
      Pbuf[wave][(r + half * 8) * 32 + 16 + col] = (bf16_t)p1;
    }

    // O += P @ V  (A = P 16x32 via LDS repack; B from transposed V tile)
    const v16bf pa = load_afrag(&Pbuf[wave][0], 32, lane);
#pragma unroll
    for (int dt = 0; dt < 4; ++dt) {
      const v16bf bvf = load_bfrag_nk(&VT[wave][dt * 16 * 33], 33, lane);
      o[dt] = wmma_bf16(pa, bvf, o[dt]);
    }
  }

  // normalize and write ctx in [B,S,E] fp32 (E index = h*64 + d)
#pragma unroll
  for (int dt = 0; dt < 4; ++dt) {
#pragma unroll
    for (int r = 0; r < 8; ++r) {
      const int m = q0 + r + half * 8;
      const int e = hh * 64 + dt * 16 + col;
      ctx[((size_t)bb * S_DIM + m) * E_DIM + e] = o[dt][r] / lrun[r];
    }
  }
}

// ---------------- Row LayerNorm (E=2048), output bf16 ----------------
__global__ __launch_bounds__(256) void ln_kernel(const float* __restrict__ ctx,
                                                 const float* __restrict__ gamma,
                                                 const float* __restrict__ beta,
                                                 bf16_t* __restrict__ Y) {
  const int row = blockIdx.x, tid = threadIdx.x;
  const float* x = ctx + (size_t)row * E_DIM;
  float s = 0.f, ss = 0.f;
  for (int i = tid; i < E_DIM; i += 256) { const float v = x[i]; s += v; ss += v * v; }
  __shared__ float rs[256], rss[256];
  rs[tid] = s; rss[tid] = ss;
  __syncthreads();
  for (int st = 128; st > 0; st >>= 1) {
    if (tid < st) { rs[tid] += rs[tid + st]; rss[tid] += rss[tid + st]; }
    __syncthreads();
  }
  const float mu  = rs[0] * (1.0f / E_DIM);
  const float var = rss[0] * (1.0f / E_DIM) - mu * mu;
  const float inv = rsqrtf(var + 1e-5f);
  for (int i = tid; i < E_DIM; i += 256) {
    const float v = (x[i] - mu) * inv * gamma[i] + beta[i];
    Y[(size_t)row * E_DIM + i] = (bf16_t)v;
  }
}

// ---------------- host launch ----------------
extern "C" void kernel_launch(void* const* d_in, const int* in_sizes, int n_in,
                              void* d_out, int out_size, void* d_ws, size_t ws_size,
                              hipStream_t stream) {
  const float* hidden = (const float*)d_in[0];
  // d_in[1] (attention_mask) is the analytic causal mask; applied in-kernel.
  const float* Wq = (const float*)d_in[2];
  const float* bq = (const float*)d_in[3];
  const float* Wk = (const float*)d_in[4];
  const float* bk = (const float*)d_in[5];
  const float* Wv = (const float*)d_in[6];
  const float* bv = (const float*)d_in[7];
  const float* Wo = (const float*)d_in[8];
  const float* bo = (const float*)d_in[9];
  const float* gamma = (const float*)d_in[10];
  const float* beta  = (const float*)d_in[11];
  float* out = (float*)d_out;

  char* ws = (char*)d_ws;
  const size_t XB = (size_t)M_DIM * E_DIM * sizeof(bf16_t);   // 16 MB
  const size_t WB = (size_t)E_DIM * E_DIM * sizeof(bf16_t);   // 8 MB
  bf16_t* Xbf = (bf16_t*)(ws);
  bf16_t* Wqb = (bf16_t*)(ws + XB);
  bf16_t* Wkb = (bf16_t*)(ws + XB + 1 * WB);
  bf16_t* Wvb = (bf16_t*)(ws + XB + 2 * WB);
  bf16_t* Wob = (bf16_t*)(ws + XB + 3 * WB);
  bf16_t* Qb  = (bf16_t*)(ws + XB + 4 * WB);
  bf16_t* Kb  = (bf16_t*)(ws + XB + 4 * WB + 1 * XB);
  bf16_t* Vb  = (bf16_t*)(ws + XB + 4 * WB + 2 * XB);
  float*  ctx = (float* )(ws + XB + 4 * WB + 3 * XB);
  bf16_t* Y   = Xbf;  // reuse X region after QKV projections are done

  const int nX = M_DIM * E_DIM;   // 8,388,608
  const int nW = E_DIM * E_DIM;   // 4,194,304
  cvt_kernel<<<nX / 1024, 256, 0, stream>>>(hidden, Xbf, nX);
  cvt_kernel<<<nW / 1024, 256, 0, stream>>>(Wq, Wqb, nW);
  cvt_kernel<<<nW / 1024, 256, 0, stream>>>(Wk, Wkb, nW);
  cvt_kernel<<<nW / 1024, 256, 0, stream>>>(Wv, Wvb, nW);
  cvt_kernel<<<nW / 1024, 256, 0, stream>>>(Wo, Wob, nW);

  const dim3 ggrid(E_DIM / 128, M_DIM / 128);
  gemm_kernel<1><<<ggrid, 256, 0, stream>>>(Xbf, Wqb, bq, Q_SCALE, (void*)Qb);
  gemm_kernel<0><<<ggrid, 256, 0, stream>>>(Xbf, Wkb, bk, 1.0f,    (void*)Kb);
  gemm_kernel<0><<<ggrid, 256, 0, stream>>>(Xbf, Wvb, bv, 1.0f,    (void*)Vb);

  attn_kernel<<<dim3(S_DIM / 128, H_DIM, B_DIM), 256, 0, stream>>>(Qb, Kb, Vb, ctx);

  ln_kernel<<<M_DIM, 256, 0, stream>>>(ctx, gamma, beta, Y);

  gemm_kernel<2><<<ggrid, 256, 0, stream>>>(Y, Wob, bo, 1.0f, (void*)out);
}
